// AugmentedMamba3_82832739270873
// MI455X (gfx1250) — matrix operand order repaired and checked by reading, not compiled, not run on hardware
//
#include <hip/hip_runtime.h>
#include <hip/hip_bf16.h>
#include <math.h>

// ---------------------------------------------------------------------------
// AugmentedMamba3 for gfx1250 (MI455X).
//   B=4, L=2048, D=1024, NREG=8, NMEM=16, DECAY=0.995, SHARP=5
//
//   1) one fused bf16 WMMA GEMM for [rv|rq|mv|mq]   (M=8192,N=4096,K=1024)
//      + GEMM for `pre` (W_h slice of comb_w)        -> 86 GFLOP total
//      Tiles staged into LDS by the Tensor Data Mover (tensor_load_to_lds),
//      double-buffered (K-loop unrolled x2 so buffer indices are constants),
//      one barrier + s_wait_tensorcnt per K-step.
//   2) tiny gate/addr kernel (sigmoid/softmax)
//   3) register-resident sequential scan (1 block/batch, 1024 thr, wave32
//      shfl reductions) -> writes [r_read|m_read] bf16
//   4) combine GEMM: [r|m](8192x2048) @ [W_r|W_m]^T  (K=2048 bf16 WMMA)
//   5) fused LayerNorm epilogue: LN(pre + comb + u)
// ---------------------------------------------------------------------------

typedef __bf16 bf16_t;
typedef __attribute__((ext_vector_type(16))) __bf16 v16bf;
typedef __attribute__((ext_vector_type(8)))  __bf16 v8bf;
typedef __attribute__((ext_vector_type(8)))  float  v8f;
typedef __attribute__((ext_vector_type(4)))  unsigned int u32x4;
typedef __attribute__((ext_vector_type(4)))  int   i32x4;
typedef __attribute__((ext_vector_type(8)))  int   i32x8;

#define D_DIM   1024
#define L_DIM   2048
#define B_DIM   4
#define TOKS    (B_DIM * L_DIM)   // 8192
#define NREG    8
#define NMEM    16
#define DECAY   0.995f
#define SHARP   5.0f

#define KSTEP   32
#define LDS_PAD 40                // bf16 elems per LDS row (32 data + 8 pad)
#define TM      128
#define TN      128

#if defined(__has_builtin)
#if __has_builtin(__builtin_amdgcn_tensor_load_to_lds) && \
    __has_builtin(__builtin_amdgcn_s_wait_tensorcnt)
#define USE_TDM 1
#endif
#endif

// ---------------------------------------------------------------------------
// f32 -> bf16 conversion
// ---------------------------------------------------------------------------
__global__ __launch_bounds__(256) void k_cvt_bf16(const float* __restrict__ src,
                                                  bf16_t* __restrict__ dst, int n) {
  int i = blockIdx.x * 256 + threadIdx.x;
  if (i < n) dst[i] = (bf16_t)src[i];
}

// ---------------------------------------------------------------------------
// WMMA fragment loads from padded LDS rows (CDNA5 ISA 7.12.2 layouts, wave32)
//   A 16x32 bf16: lane<16 row M=lane holds K {0..7,16..23}; lane>=16 the rest
//   B 32x16 bf16: lane<16 col N=lane holds K {0..7,16..23}; lane>=16 the rest
// ---------------------------------------------------------------------------
__device__ __forceinline__ v16bf load_a_frag(const bf16_t* row, int half) {
  union { v16bf v; v8bf h[2]; } u;
  u.h[0] = *(const v8bf*)&row[half * 8];
  u.h[1] = *(const v8bf*)&row[16 + half * 8];
  return u.v;
}
__device__ __forceinline__ v16bf load_b_frag(const bf16_t* row, int half) {
  union { v16bf v; v8bf h[2]; } u;
  u.h[0] = *(const v8bf*)&row[half * 8];
  u.h[1] = *(const v8bf*)&row[16 + half * 8];
  return u.v;
}

// 8 WMMAs of one K-step for this wave's 32x64 sub-tile
__device__ __forceinline__ void mma_step(const bf16_t* AsB, const bf16_t* BsB,
                                         int mw, int nw, int l16, int half,
                                         v8f acc[2][4]) {
  v16bf af[2], bf[4];
#pragma unroll
  for (int r = 0; r < 2; ++r)
    af[r] = load_a_frag(&AsB[(mw + r * 16 + l16) * LDS_PAD], half);
#pragma unroll
  for (int c = 0; c < 4; ++c)
    bf[c] = load_b_frag(&BsB[(nw + c * 16 + l16) * LDS_PAD], half);
#pragma unroll
  for (int r = 0; r < 2; ++r)
#pragma unroll
    for (int c = 0; c < 4; ++c)
      acc[r][c] = __builtin_amdgcn_wmma_f32_16x16x32_bf16(
          false, af[r], false, bf[c], (short)0, acc[r][c], false, false);
}

#ifdef USE_TDM
// ---------------------------------------------------------------------------
// Issue one TDM 2D tile load: TM rows x KSTEP bf16, row stride `ld` elements,
// into LDS at byte offset lds_off with 16B padding after each 64B row
// (pad_interval=3 -> every 16 DWORDs, pad_amount=3 -> 4 DWORDs) so the LDS
// row stride is exactly LDS_PAD(=40) bf16 elements.
// ---------------------------------------------------------------------------
__device__ __forceinline__ void tdm_load_tile(const bf16_t* gptr, unsigned lds_off,
                                              int ld) {
  const unsigned long long ga = (unsigned long long)(uintptr_t)gptr;
  u32x4 g0;
  g0[0] = 1u;                                       // count=1, user descriptor
  g0[1] = lds_off;                                  // LDS byte address
  g0[2] = (unsigned)ga;                             // global_addr[31:0]
  g0[3] = (unsigned)((ga >> 32) & 0x01FFFFFFu) | (2u << 30);  // addr hi | type=2

  const unsigned dim0 = 1u << 20;                   // huge dims: never OOB-clip
  const unsigned dim1 = 1u << 20;
  i32x8 g1;
  g1[0] = (int)((1u << 16)      // data_size = 1 (2 bytes)
              | (1u << 20)      // pad_enable
              | (3u << 22)      // pad_interval: 16 DWORDs (=64B row)
              | (3u << 25));    // pad_amount:   4 DWORDs (=16B pad)
  g1[1] = (int)((dim0 & 0xFFFFu) << 16);            // tensor_dim0[15:0]
  g1[2] = (int)((dim0 >> 16) | ((dim1 & 0xFFFFu) << 16));
  g1[3] = (int)((dim1 >> 16) | ((unsigned)KSTEP << 16));   // tile_dim0 = 32
  g1[4] = (int)((unsigned)TM);                      // tile_dim1 = 128, tile_dim2=0
  g1[5] = (int)(unsigned)ld;                        // tensor_dim0_stride lo
  g1[6] = 0;
  g1[7] = 0;
  i32x4 z4 = {0, 0, 0, 0};
#if __clang_major__ >= 23
  i32x8 z8 = {0, 0, 0, 0, 0, 0, 0, 0};
  __builtin_amdgcn_tensor_load_to_lds(g0, g1, z4, z4, z8, 0);
#else
  __builtin_amdgcn_tensor_load_to_lds(g0, g1, z4, z4, 0);
#endif
}
#endif  // USE_TDM

// ---------------------------------------------------------------------------
// Tiled bf16 WMMA GEMM:  C[m][n] = sum_k A[m][k] * W[n][k]  (+ bias[n])
// Block: 256 threads (8 waves). Tile: 128(M) x 128(N), K-step 32.
// Wave w: m-sub (w&3)*32 (2 A-frags), n-sub (w>>2)*64 (4 B-frags)
//   -> 8 WMMAs per wave per K-step against 12 ds_load_b128.
// K must be a multiple of 2*KSTEP (holds: K = 1024 or 2048).
// ---------------------------------------------------------------------------
__global__ __launch_bounds__(256) void k_gemm_bf16(
    const bf16_t* __restrict__ A, int lda,
    const bf16_t* __restrict__ W, int ldb,
    const float* __restrict__ bias,
    float* __restrict__ C, int ldc, int K) {
  const int m0 = blockIdx.x * TM;
  const int n0 = blockIdx.y * TN;
  const int t  = threadIdx.x;
  const int wave = t >> 5;
  const int lane = t & 31;
  const int half = lane >> 4;
  const int l16  = lane & 15;
  const int mw   = (wave & 3) << 5;   // 0,32,64,96
  const int nw   = (wave >> 2) << 6;  // 0,64

  v8f acc[2][4] = {};

#ifdef USE_TDM
  // -------- TDM double-buffered staging; K-loop unrolled x2 so each half
  // addresses a fixed LDS buffer (descriptor build stays pure SALU).
  __shared__ bf16_t As[2][TM * LDS_PAD];
  __shared__ bf16_t Bs[2][TM * LDS_PAD];
  const bf16_t* Abase = A + (size_t)m0 * lda;
  const bf16_t* Wbase = W + (size_t)n0 * ldb;
  const unsigned ldsA0 = (unsigned)(uintptr_t)&As[0][0];
  const unsigned ldsA1 = (unsigned)(uintptr_t)&As[1][0];
  const unsigned ldsB0 = (unsigned)(uintptr_t)&Bs[0][0];
  const unsigned ldsB1 = (unsigned)(uintptr_t)&Bs[1][0];

  if (wave == 0) {                      // prologue: DMA tile 0 into buffer 0
    tdm_load_tile(Abase, ldsA0, lda);
    tdm_load_tile(Wbase, ldsB0, ldb);
  }
  for (int k0 = 0; k0 < K; k0 += 2 * KSTEP) {
    // ---- half-step A: compute on buffer 0, DMA k0+KSTEP into buffer 1
    __builtin_amdgcn_s_wait_tensorcnt(0);   // issuer waits; others: count==0
    __syncthreads();                        // buffer 0 visible to all waves
    if (wave == 0) {                        // k0+KSTEP < K always (even #steps)
      tdm_load_tile(Abase + k0 + KSTEP, ldsA1, lda);
      tdm_load_tile(Wbase + k0 + KSTEP, ldsB1, ldb);
    }
    mma_step(As[0], Bs[0], mw, nw, l16, half, acc);

    // ---- half-step B: compute on buffer 1, DMA k0+2*KSTEP into buffer 0
    __builtin_amdgcn_s_wait_tensorcnt(0);
    __syncthreads();                        // buffer 1 visible; buffer-0 reads
                                            // done (dscnt waited before WMMAs)
    if (wave == 0 && k0 + 2 * KSTEP < K) {
      tdm_load_tile(Abase + k0 + 2 * KSTEP, ldsA0, lda);
      tdm_load_tile(Wbase + k0 + 2 * KSTEP, ldsB0, ldb);
    }
    mma_step(As[1], Bs[1], mw, nw, l16, half, acc);
  }
#else
  // -------- fallback: manual global->LDS staging, single buffer
  __shared__ bf16_t As[TM * LDS_PAD];
  __shared__ bf16_t Bs[TM * LDS_PAD];
  const int lrow = t >> 2;           // 0..63 (plus +64 row)
  const int lcol = (t & 3) << 3;     // 0,8,16,24
  const size_t arow0 = (size_t)(m0 + lrow) * lda;
  const size_t arow1 = (size_t)(m0 + lrow + 64) * lda;
  const size_t brow0 = (size_t)(n0 + lrow) * ldb;
  const size_t brow1 = (size_t)(n0 + lrow + 64) * ldb;
  for (int k0 = 0; k0 < K; k0 += KSTEP) {
    *(uint4*)&As[lrow * LDS_PAD + lcol]        = *(const uint4*)&A[arow0 + k0 + lcol];
    *(uint4*)&As[(lrow + 64) * LDS_PAD + lcol] = *(const uint4*)&A[arow1 + k0 + lcol];
    *(uint4*)&Bs[lrow * LDS_PAD + lcol]        = *(const uint4*)&W[brow0 + k0 + lcol];
    *(uint4*)&Bs[(lrow + 64) * LDS_PAD + lcol] = *(const uint4*)&W[brow1 + k0 + lcol];
    __syncthreads();
    mma_step(As, Bs, mw, nw, l16, half, acc);
    __syncthreads();
  }
#endif

  // -------- epilogue: C/D layout VGPR j -> row M = j + half*8, col N = l16
#pragma unroll
  for (int r = 0; r < 2; ++r) {
    const int crow = m0 + mw + r * 16 + half * 8;
#pragma unroll
    for (int c = 0; c < 4; ++c) {
      const int ccol = n0 + nw + c * 16 + l16;
      const float bv = bias ? bias[ccol] : 0.0f;
#pragma unroll
      for (int j = 0; j < 8; ++j)
        C[(size_t)(crow + j) * ldc + ccol] = acc[r][c][j] + bv;
    }
  }
}

// ---------------------------------------------------------------------------
// Gates + addresses: one wave per token.
// ---------------------------------------------------------------------------
__global__ __launch_bounds__(256) void k_gates(
    const float* __restrict__ u,
    const float* __restrict__ rgw, const float* __restrict__ rgb,
    const float* __restrict__ raw, const float* __restrict__ rab,
    const float* __restrict__ mgw, const float* __restrict__ mgb,
    const float* __restrict__ maw, const float* __restrict__ mab,
    float* __restrict__ rg, float* __restrict__ ra,
    float* __restrict__ mg, float* __restrict__ ma) {
  const int lane = threadIdx.x & 31;
  const int wave = threadIdx.x >> 5;
  const int tok  = blockIdx.x * 8 + wave;
  const float* up = u + (size_t)tok * D_DIM;

  float acc[26];
#pragma unroll
  for (int i = 0; i < 26; ++i) acc[i] = 0.0f;

  for (int j = 0; j < D_DIM / 32; ++j) {
    const int d = j * 32 + lane;
    const float uv = up[d];
    acc[0] += uv * rgw[d];
#pragma unroll
    for (int n = 0; n < NREG; ++n) acc[1 + n] += uv * raw[n * D_DIM + d];
    acc[9] += uv * mgw[d];
#pragma unroll
    for (int n = 0; n < NMEM; ++n) acc[10 + n] += uv * maw[n * D_DIM + d];
  }
#pragma unroll
  for (int i = 0; i < 26; ++i) {
#pragma unroll
    for (int off = 16; off >= 1; off >>= 1)
      acc[i] += __shfl_xor(acc[i], off, 32);
  }
  if (lane == 0) {
    rg[tok] = 1.0f / (1.0f + __expf(-SHARP * (acc[0] + rgb[0])));
    mg[tok] = 1.0f / (1.0f + __expf(-SHARP * (acc[9] + mgb[0])));
  }
  {
    float mx = -1e30f;
#pragma unroll
    for (int n = 0; n < NREG; ++n) mx = fmaxf(mx, acc[1 + n] + rab[n]);
    float sum = 0.0f, mine = 0.0f;
#pragma unroll
    for (int n = 0; n < NREG; ++n) {
      float e = __expf(acc[1 + n] + rab[n] - mx);
      sum += e;
      if (lane == n) mine = e;
    }
    if (lane < NREG) ra[(size_t)tok * NREG + lane] = mine / sum;
  }
  {
    float mx = -1e30f;
#pragma unroll
    for (int n = 0; n < NMEM; ++n) mx = fmaxf(mx, acc[10 + n] + mab[n]);
    float sum = 0.0f, mine = 0.0f;
#pragma unroll
    for (int n = 0; n < NMEM; ++n) {
      float e = __expf(acc[10 + n] + mab[n] - mx);
      sum += e;
      if (lane == n) mine = e;
    }
    if (lane < NMEM) ma[(size_t)tok * NMEM + lane] = mine / sum;
  }
}

// ---------------------------------------------------------------------------
// Sequential scan: grid = B, 1024 threads; thread d owns channel d; state
// reg_s[8]+mem_s[16] in VGPRs. P holds fused [rv|rq|mv|mq] rows (stride 4096);
// channel biases folded in here (loaded once, kept in registers).
// ---------------------------------------------------------------------------
__global__ __launch_bounds__(1024) void k_scan(
    const float* __restrict__ P,
    const float* __restrict__ rvb, const float* __restrict__ rqb,
    const float* __restrict__ mvb, const float* __restrict__ mqb,
    const float* __restrict__ rg, const float* __restrict__ ra,
    const float* __restrict__ mg, const float* __restrict__ ma,
    bf16_t* __restrict__ RM) {
  const int b    = blockIdx.x;
  const int d    = threadIdx.x;
  const int lane = d & 31;
  const int wave = d >> 5;

  __shared__ float red[32][25];
  __shared__ float scores[24];

  float regs[NREG];
  float mems[NMEM];
#pragma unroll
  for (int n = 0; n < NREG; ++n) regs[n] = 0.0f;
#pragma unroll
  for (int n = 0; n < NMEM; ++n) mems[n] = 0.0f;

  const float rvb_d = rvb[d], rqb_d = rqb[d], mvb_d = mvb[d], mqb_d = mqb[d];
  const float scale = 0.03125f;   // D^-0.5

  for (int t = 0; t < L_DIM; ++t) {
    const size_t tok  = (size_t)b * L_DIM + t;
    const size_t base = tok * (4 * D_DIM) + d;
    const float rv_d = P[base]             + rvb_d;
    const float rq_d = P[base + D_DIM]     + rqb_d;
    const float mv_d = P[base + 2 * D_DIM] + mvb_d;
    const float mq_d = P[base + 3 * D_DIM] + mqb_d;
    const float rg_t = rg[tok], mg_t = mg[tok];

    if (lane == 0 && t + 1 < L_DIM) {   // pull next step's rows toward L2/L0
      const float* nx = &P[(tok + 1) * (4 * D_DIM) + wave * 32];
#pragma unroll
      for (int p = 0; p < 4; ++p) __builtin_prefetch(nx + p * D_DIM, 0, 1);
    }

    float pr[NREG];
#pragma unroll
    for (int n = 0; n < NREG; ++n) {
      regs[n] += rg_t * ra[tok * NREG + n] * rv_d;
      pr[n] = regs[n] * rq_d;
    }
    float pm[NMEM];
#pragma unroll
    for (int n = 0; n < NMEM; ++n) {
      mems[n] = mems[n] * DECAY + mg_t * ma[tok * NMEM + n] * mv_d;
      pm[n] = mems[n] * mq_d;
    }
#pragma unroll
    for (int n = 0; n < NREG; ++n)
#pragma unroll
      for (int off = 16; off >= 1; off >>= 1) pr[n] += __shfl_xor(pr[n], off, 32);
#pragma unroll
    for (int n = 0; n < NMEM; ++n)
#pragma unroll
      for (int off = 16; off >= 1; off >>= 1) pm[n] += __shfl_xor(pm[n], off, 32);
    if (lane == 0) {
#pragma unroll
      for (int n = 0; n < NREG; ++n) red[wave][n] = pr[n];
#pragma unroll
      for (int n = 0; n < NMEM; ++n) red[wave][NREG + n] = pm[n];
    }
    __syncthreads();
    if (d < 24) {
      float s = 0.0f;
#pragma unroll
      for (int w2 = 0; w2 < 32; ++w2) s += red[w2][d];
      scores[d] = s * scale;
    }
    __syncthreads();

    float r_read, m_read;
    {
      float mx = scores[0];
#pragma unroll
      for (int n = 1; n < NREG; ++n) mx = fmaxf(mx, scores[n]);
      float sum = 0.0f, accv = 0.0f;
#pragma unroll
      for (int n = 0; n < NREG; ++n) {
        float e = __expf(scores[n] - mx);
        sum  += e;
        accv += e * regs[n];
      }
      r_read = accv / sum;
    }
    {
      float mx = scores[NREG];
#pragma unroll
      for (int n = 1; n < NMEM; ++n) mx = fmaxf(mx, scores[NREG + n]);
      float sum = 0.0f, accv = 0.0f;
#pragma unroll
      for (int n = 0; n < NMEM; ++n) {
        float e = __expf(scores[NREG + n] - mx);
        sum  += e;
        accv += e * mems[n];
      }
      m_read = accv / sum;
    }
    RM[tok * (2 * D_DIM) + d]         = (bf16_t)r_read;
    RM[tok * (2 * D_DIM) + D_DIM + d] = (bf16_t)m_read;
  }
}

// ---------------------------------------------------------------------------
// Epilogue: out = LN(pre + cmb + u)
// ---------------------------------------------------------------------------
__global__ __launch_bounds__(256) void k_ln_out(
    const float* __restrict__ pre, const float* __restrict__ cmb,
    const float* __restrict__ u,
    const float* __restrict__ lng, const float* __restrict__ lnb,
    float* __restrict__ out) {
  const int tok = blockIdx.x;
  const size_t base = (size_t)tok * D_DIM;
  const int lane = threadIdx.x & 31;
  const int wave = threadIdx.x >> 5;

  float x[4];
  float s = 0.0f, s2 = 0.0f;
#pragma unroll
  for (int j = 0; j < 4; ++j) {
    const int idx = threadIdx.x + j * 256;
    const float v = pre[base + idx] + cmb[base + idx] + u[base + idx];
    x[j] = v;
    s += v;
    s2 += v * v;
  }
#pragma unroll
  for (int off = 16; off >= 1; off >>= 1) {
    s  += __shfl_xor(s, off, 32);
    s2 += __shfl_xor(s2, off, 32);
  }
  __shared__ float sw[8], s2w[8];
  if (lane == 0) { sw[wave] = s; s2w[wave] = s2; }
  __syncthreads();
  if (threadIdx.x == 0) {
    float ts = 0.0f, t2 = 0.0f;
#pragma unroll
    for (int w2 = 0; w2 < 8; ++w2) { ts += sw[w2]; t2 += s2w[w2]; }
    sw[0]  = ts * (1.0f / D_DIM);
    s2w[0] = t2 * (1.0f / D_DIM);
  }
  __syncthreads();
  const float mean = sw[0];
  const float var  = s2w[0] - mean * mean;
  const float inv  = rsqrtf(var + 1e-5f);
#pragma unroll
  for (int j = 0; j < 4; ++j) {
    const int idx = threadIdx.x + j * 256;
    out[base + idx] = lng[idx] * (x[j] - mean) * inv + lnb[idx];
  }
}

// ---------------------------------------------------------------------------
// Host orchestration
// ---------------------------------------------------------------------------
extern "C" void kernel_launch(void* const* d_in, const int* in_sizes, int n_in,
                              void* d_out, int out_size, void* d_ws, size_t ws_size,
                              hipStream_t stream) {
  const float* u     = (const float*)d_in[0];
  const float* rgw   = (const float*)d_in[1];
  const float* rgb   = (const float*)d_in[2];
  const float* raw_  = (const float*)d_in[3];
  const float* rab   = (const float*)d_in[4];
  const float* rvw   = (const float*)d_in[5];
  const float* rvb   = (const float*)d_in[6];
  const float* rqw   = (const float*)d_in[7];
  const float* rqb   = (const float*)d_in[8];
  const float* mgw   = (const float*)d_in[9];
  const float* mgb   = (const float*)d_in[10];
  const float* maw_  = (const float*)d_in[11];
  const float* mab   = (const float*)d_in[12];
  const float* mvw   = (const float*)d_in[13];
  const float* mvb   = (const float*)d_in[14];
  const float* mqw   = (const float*)d_in[15];
  const float* mqb   = (const float*)d_in[16];
  const float* combw = (const float*)d_in[17];
  const float* combb = (const float*)d_in[18];
  const float* lng   = (const float*)d_in[19];
  const float* lnb   = (const float*)d_in[20];
  float* out = (float*)d_out;

  // ---- workspace carve-out ----
  char* base = (char*)d_ws;
  size_t off = 0;
  auto take = [&](size_t bytes) -> char* {
    char* p = base + off;
    off += (bytes + 255) & ~(size_t)255;
    return p;
  };
  bf16_t* u_bf    = (bf16_t*)take((size_t)TOKS * D_DIM * 2);              // 16 MB
  bf16_t* wproj   = (bf16_t*)take((size_t)4 * D_DIM * D_DIM * 2);         // 8 MB: [rv|rq|mv|mq] rows
  bf16_t* comb_bf = (bf16_t*)take((size_t)D_DIM * 3 * D_DIM * 2);         // 6 MB
  float*  P       = (float*)take((size_t)TOKS * 4 * D_DIM * 4);           // 128 MB fused proj out
  float*  pre     = (float*)take((size_t)TOKS * D_DIM * 4);               // 32 MB
  float*  rg      = (float*)take((size_t)TOKS * 4);
  float*  ra      = (float*)take((size_t)TOKS * NREG * 4);
  float*  mg      = (float*)take((size_t)TOKS * 4);
  float*  ma      = (float*)take((size_t)TOKS * NMEM * 4);
  bf16_t* RM      = (bf16_t*)take((size_t)TOKS * 2 * D_DIM * 2);          // 32 MB
  float*  cmb     = P;   // P dead after the scan; reuse for combine GEMM out

  // ---- 1) bf16 conversions (weights stacked: rows 0..1023=rv, ..=rq,mv,mq)
  const int nU = TOKS * D_DIM;
  const int nW = D_DIM * D_DIM;
  const int nC = D_DIM * 3 * D_DIM;
  k_cvt_bf16<<<(nU + 255) / 256, 256, 0, stream>>>(u,     u_bf,              nU);
  k_cvt_bf16<<<(nW + 255) / 256, 256, 0, stream>>>(rvw,   wproj + 0 * nW,    nW);
  k_cvt_bf16<<<(nW + 255) / 256, 256, 0, stream>>>(rqw,   wproj + 1 * (size_t)nW, nW);
  k_cvt_bf16<<<(nW + 255) / 256, 256, 0, stream>>>(mvw,   wproj + 2 * (size_t)nW, nW);
  k_cvt_bf16<<<(nW + 255) / 256, 256, 0, stream>>>(mqw,   wproj + 3 * (size_t)nW, nW);
  k_cvt_bf16<<<(nC + 255) / 256, 256, 0, stream>>>(combw, comb_bf,           nC);

  // ---- 2) projection GEMMs (WMMA bf16, TDM-staged) ----
  // fused [rv|rq|mv|mq]: M=8192, N=4096, K=1024 (biases folded into scan)
  dim3 gproj(TOKS / TM, (4 * D_DIM) / TN);   // 64 x 32
  k_gemm_bf16<<<gproj, 256, 0, stream>>>(u_bf, D_DIM, wproj, D_DIM,
                                         nullptr, P, 4 * D_DIM, D_DIM);
  // pre = u @ W_h^T + comb_b ; W_h rows at comb_bf[j*3072 + 0..1023]
  dim3 gpre(TOKS / TM, D_DIM / TN);          // 64 x 8
  k_gemm_bf16<<<gpre, 256, 0, stream>>>(u_bf, D_DIM, comb_bf, 3 * D_DIM,
                                        combb, pre, D_DIM, D_DIM);

  // ---- 3) gates / addresses ----
  k_gates<<<TOKS / 8, 256, 0, stream>>>(u, rgw, rgb, raw_, rab, mgw, mgb, maw_, mab,
                                        rg, ra, mg, ma);

  // ---- 4) sequential scan ----
  k_scan<<<B_DIM, 1024, 0, stream>>>(P, rvb, rqb, mvb, mqb,
                                     rg, ra, mg, ma, RM);

  // ---- 5) combine GEMM: [r|m] @ [W_r|W_m]^T, K=2048 ----
  k_gemm_bf16<<<gpre, 256, 0, stream>>>(RM, 2 * D_DIM, comb_bf + D_DIM, 3 * D_DIM,
                                        nullptr, cmb, D_DIM, 2 * D_DIM);

  // ---- 6) fused LayerNorm epilogue ----
  k_ln_out<<<TOKS, 256, 0, stream>>>(pre, cmb, u, lng, lnb, out);

  (void)in_sizes; (void)n_in; (void)out_size; (void)ws_size;
}
